// Encoder_14800457302527
// MI455X (gfx1250) — matrix operand order repaired and checked by reading, not compiled, not run on hardware
//
#include <hip/hip_runtime.h>
#include <hip/hip_bf16.h>

typedef __attribute__((ext_vector_type(16))) __bf16 bf16x16;
typedef __attribute__((ext_vector_type(8)))  __bf16 bf16x8;
typedef __attribute__((ext_vector_type(8)))  float  f32x8;

#define TB 256

// ---------------- degree / normalization ----------------

__global__ void deg_init_kernel(float* __restrict__ deg, int n) {
    int i = blockIdx.x * blockDim.x + threadIdx.x;
    if (i < n) deg[i] = 1.0f;   // self-loop contributes 1
}

__global__ void deg_edge_kernel(const int* __restrict__ dst, float* __restrict__ deg, int e) {
    int i = blockIdx.x * blockDim.x + threadIdx.x;
    if (i < e) atomicAdd(&deg[dst[i]], 1.0f);
}

__global__ void dinv_kernel(const float* __restrict__ deg, float* __restrict__ dinv, int n) {
    int i = blockIdx.x * blockDim.x + threadIdx.x;
    if (i < n) dinv[i] = rsqrtf(deg[i]);   // deg >= 1 always (self loops)
}

// ---------------- conversions / packing ----------------

__global__ void cvt_bf16_kernel(const float* __restrict__ in, __bf16* __restrict__ out, size_t n) {
    size_t i = (size_t)blockIdx.x * blockDim.x + threadIdx.x;
    if (i < n) out[i] = (__bf16)in[i];
}

// Pack W [K,N] f32 row-major into bf16 WMMA B-fragment tiles (32x16 per tile).
// Flat layout: tile = kt*(N/16)+nt ; within tile: lane*16 + e
// element e of lane's v16bf holds B[k = kt*32 + (lane>=16?8:0) + (e&7) + (e>=8?16:0)][n = nt*16 + (lane&15)]
__global__ void pack_b_kernel(const float* __restrict__ W, __bf16* __restrict__ Bp, int K, int N) {
    int idx = blockIdx.x * blockDim.x + threadIdx.x;
    if (idx >= K * N) return;
    int tile   = idx >> 9;        // 512 elements per 32x16 tile
    int within = idx & 511;
    int lane   = within >> 4;
    int e      = within & 15;
    int ntiles = N >> 4;
    int kt = tile / ntiles;
    int nt = tile - kt * ntiles;
    int k = kt * 32 + (lane >> 4) * 8 + (e & 7) + (e >> 3) * 16;
    int n = nt * 16 + (lane & 15);
    Bp[idx] = (__bf16)W[(size_t)k * N + n];
}

// ---------------- WMMA GEMM: C[M,N] = A[M,K] @ B[K,N] (+bias) ----------------

__device__ inline bf16x16 load_a_frag(const __bf16* __restrict__ A, int M, int K,
                                      int m0, int k0, int lane) {
    // 16-bit A 16x32 layout: lanes 0-15 -> K {0..7,16..23}, lanes 16-31 -> K {8..15,24..31}
    int half = lane >> 4;
    int r = m0 + (lane & 15);
    if (r >= M) r = M - 1;              // clamp; store side is guarded
    const __bf16* p = A + (size_t)r * K + k0 + half * 8;
    bf16x8 lo = *(const bf16x8*)(p);
    bf16x8 hi = *(const bf16x8*)(p + 16);
    bf16x16 a;
#pragma unroll
    for (int i = 0; i < 8; ++i) { a[i] = lo[i]; a[i + 8] = hi[i]; }
    return a;
}

template <int BIAS, int OUTBF>
__global__ void gemm_wmma_bf16(const __bf16* __restrict__ A,
                               const __bf16* __restrict__ Bp,   // packed fragments
                               const float*  __restrict__ bias,
                               float*  __restrict__ Cf,
                               __bf16* __restrict__ Cb,
                               int M, int N, int K) {
    const int lane = threadIdx.x & 31;
    const int w    = threadIdx.x >> 5;      // 8 waves
    const int wm   = w >> 2;                // 0..1
    const int wn   = w & 3;                 // 0..3
    const int m0 = blockIdx.x * 64 + wm * 32;
    const int n0 = blockIdx.y * 128 + wn * 32;
    const int ntiles = N >> 4;

    f32x8 acc00 = {}, acc01 = {}, acc10 = {}, acc11 = {};

    for (int k0 = 0; k0 < K; k0 += 32) {
        bf16x16 a0 = load_a_frag(A, M, K, m0,      k0, lane);
        bf16x16 a1 = load_a_frag(A, M, K, m0 + 16, k0, lane);
        const __bf16* bb = Bp + ((size_t)((k0 >> 5) * ntiles + (n0 >> 4))) * 512 + lane * 16;
        bf16x16 b0 = *(const bf16x16*)(bb);
        bf16x16 b1 = *(const bf16x16*)(bb + 512);

        acc00 = __builtin_amdgcn_wmma_f32_16x16x32_bf16(false, a0, false, b0, (short)0, acc00, false, false);
        acc01 = __builtin_amdgcn_wmma_f32_16x16x32_bf16(false, a0, false, b1, (short)0, acc01, false, false);
        acc10 = __builtin_amdgcn_wmma_f32_16x16x32_bf16(false, a1, false, b0, (short)0, acc10, false, false);
        acc11 = __builtin_amdgcn_wmma_f32_16x16x32_bf16(false, a1, false, b1, (short)0, acc11, false, false);
    }

    // C/D layout: VGPR i -> M = i + (lane>=16 ? 8 : 0), N = lane&15
    const int half = lane >> 4;
    const int col  = lane & 15;
    f32x8 acc[2][2] = { { acc00, acc01 }, { acc10, acc11 } };
#pragma unroll
    for (int mi = 0; mi < 2; ++mi) {
#pragma unroll
        for (int ni = 0; ni < 2; ++ni) {
            int c = n0 + ni * 16 + col;
            float bv = BIAS ? bias[c] : 0.0f;
            int rbase = m0 + mi * 16 + half * 8;
#pragma unroll
            for (int i = 0; i < 8; ++i) {
                int r = rbase + i;
                if (r < M) {
                    float v = acc[mi][ni][i] + bv;
                    if (OUTBF) Cb[(size_t)r * N + c] = (__bf16)v;
                    else       Cf[(size_t)r * N + c] = v;
                }
            }
        }
    }
}

// ---------------- graph aggregation ----------------

// self-loop init: agg[i][f] = dinv[i]^2 * hw[i][f]
__global__ void init_agg_kernel(const float* __restrict__ dinv, const float* __restrict__ hw,
                                float* __restrict__ agg, size_t total, int shiftF) {
    size_t idx = (size_t)blockIdx.x * blockDim.x + threadIdx.x;
    if (idx >= total) return;
    float di = dinv[idx >> shiftF];
    agg[idx] = di * di * hw[idx];
}

// one wave32 per edge: agg[dst] += dinv[src]*dinv[dst] * hw[src]
__global__ void scatter_kernel(const int* __restrict__ src, const int* __restrict__ dst,
                               const float* __restrict__ dinv, const float* __restrict__ hw,
                               float* __restrict__ agg, int E, int F) {
    int gw   = (int)(((size_t)blockIdx.x * blockDim.x + threadIdx.x) >> 5);
    int lane = threadIdx.x & 31;
    if (gw >= E) return;
    int s = src[gw], d = dst[gw];
    float c = dinv[s] * dinv[d];
    const float4* hs = (const float4*)(hw + (size_t)s * F);
    float* ad = agg + (size_t)d * F;
    int nvec = F >> 2;
    for (int i = lane; i < nvec; i += 32) {
        float4 v = hs[i];
        int b = i << 2;
        atomicAdd(ad + b + 0, c * v.x);
        atomicAdd(ad + b + 1, c * v.y);
        atomicAdd(ad + b + 2, c * v.z);
        atomicAdd(ad + b + 3, c * v.w);
    }
}

// ---------------- epilogues ----------------

__global__ void sigmoid_bf16_kernel(const float* __restrict__ agg, const float* __restrict__ bias,
                                    __bf16* __restrict__ out, size_t total, int maskF) {
    size_t idx = (size_t)blockIdx.x * blockDim.x + threadIdx.x;
    if (idx >= total) return;
    float v = agg[idx] + bias[idx & (size_t)maskF];
    out[idx] = (__bf16)(1.0f / (1.0f + __expf(-v)));
}

__global__ void sigmoid_f32_kernel(const float* __restrict__ agg, const float* __restrict__ bias,
                                   float* __restrict__ out, size_t total, int maskF) {
    size_t idx = (size_t)blockIdx.x * blockDim.x + threadIdx.x;
    if (idx >= total) return;
    float v = agg[idx] + bias[idx & (size_t)maskF];
    out[idx] = 1.0f / (1.0f + __expf(-v));
}

// ---------------- launch ----------------

extern "C" void kernel_launch(void* const* d_in, const int* in_sizes, int n_in,
                              void* d_out, int out_size, void* d_ws, size_t ws_size,
                              hipStream_t stream) {
    const float* x    = (const float*)d_in[0];
    const int*   ei   = (const int*)  d_in[1];   // [2,E] flat: src=ei[0..E), dst=ei[E..2E)
    const float* W_ae = (const float*)d_in[2];
    const float* b_ae = (const float*)d_in[3];
    const float* W1   = (const float*)d_in[4];
    const float* b1   = (const float*)d_in[5];
    const float* W2   = (const float*)d_in[6];
    const float* b2   = (const float*)d_in[7];

    const int IN  = 256, HID = 512, OUT = 256;
    const int Nn = in_sizes[0] / IN;
    const int E  = in_sizes[1] / 2;
    const int* src = ei;
    const int* dst = ei + E;

    // workspace carve-up
    char* p = (char*)d_ws;
    auto alloc = [&](size_t bytes) -> void* {
        void* r = (void*)p;
        p += (bytes + 255) & ~(size_t)255;
        return r;
    };
    float*  deg   = (float*) alloc((size_t)Nn * 4);
    float*  dinv  = (float*) alloc((size_t)Nn * 4);
    __bf16* xb    = (__bf16*)alloc((size_t)Nn * IN * 2);
    __bf16* Wae_p = (__bf16*)alloc((size_t)IN * HID * 2);
    __bf16* W1_p  = (__bf16*)alloc((size_t)HID * HID * 2);
    __bf16* W2_p  = (__bf16*)alloc((size_t)HID * OUT * 2);
    __bf16* hb    = (__bf16*)alloc((size_t)Nn * HID * 2);   // encoder output (bf16)
    float*  hw    = (float*) alloc((size_t)Nn * HID * 4);   // reused: hw1 then hw2
    float*  agg   = (float*) alloc((size_t)Nn * HID * 4);   // reused: agg1 then agg2
    __bf16* h2b   = (__bf16*)alloc((size_t)Nn * HID * 2);   // conv1 output (bf16)

    // 1) symmetric normalization
    deg_init_kernel<<<(Nn + TB - 1) / TB, TB, 0, stream>>>(deg, Nn);
    deg_edge_kernel<<<(E + TB - 1) / TB, TB, 0, stream>>>(dst, deg, E);
    dinv_kernel<<<(Nn + TB - 1) / TB, TB, 0, stream>>>(deg, dinv, Nn);

    // 2) convert inputs / pack weights
    cvt_bf16_kernel<<<(unsigned)(((size_t)Nn * IN + TB - 1) / TB), TB, 0, stream>>>(x, xb, (size_t)Nn * IN);
    pack_b_kernel<<<(IN * HID + TB - 1) / TB, TB, 0, stream>>>(W_ae, Wae_p, IN, HID);
    pack_b_kernel<<<(HID * HID + TB - 1) / TB, TB, 0, stream>>>(W1, W1_p, HID, HID);
    pack_b_kernel<<<(HID * OUT + TB - 1) / TB, TB, 0, stream>>>(W2, W2_p, HID, OUT);

    dim3 g1((Nn + 63) / 64, HID / 128);
    dim3 g3((Nn + 63) / 64, OUT / 128);

    // 3) encoder: h = x @ W_ae + b_ae   (bf16 out, feeds next WMMA)
    gemm_wmma_bf16<1, 1><<<g1, TB, 0, stream>>>(xb, Wae_p, b_ae, nullptr, hb, Nn, HID, IN);

    // 4) conv1: hw1 = h @ W1 ; agg = A_hat hw1 ; h2 = sigmoid(agg + b1)
    gemm_wmma_bf16<0, 0><<<g1, TB, 0, stream>>>(hb, W1_p, nullptr, hw, nullptr, Nn, HID, HID);
    init_agg_kernel<<<(unsigned)(((size_t)Nn * HID + TB - 1) / TB), TB, 0, stream>>>(dinv, hw, agg, (size_t)Nn * HID, 9);
    scatter_kernel<<<(unsigned)(((size_t)E * 32 + TB - 1) / TB), TB, 0, stream>>>(src, dst, dinv, hw, agg, E, HID);
    sigmoid_bf16_kernel<<<(unsigned)(((size_t)Nn * HID + TB - 1) / TB), TB, 0, stream>>>(agg, b1, h2b, (size_t)Nn * HID, HID - 1);

    // 5) conv2: hw2 = h2 @ W2 ; agg = A_hat hw2 ; out = sigmoid(agg + b2)
    gemm_wmma_bf16<0, 0><<<g3, TB, 0, stream>>>(h2b, W2_p, nullptr, hw, nullptr, Nn, OUT, HID);
    init_agg_kernel<<<(unsigned)(((size_t)Nn * OUT + TB - 1) / TB), TB, 0, stream>>>(dinv, hw, agg, (size_t)Nn * OUT, 8);
    scatter_kernel<<<(unsigned)(((size_t)E * 32 + TB - 1) / TB), TB, 0, stream>>>(src, dst, dinv, hw, agg, E, OUT);
    sigmoid_f32_kernel<<<(unsigned)(((size_t)Nn * OUT + TB - 1) / TB), TB, 0, stream>>>(agg, b2, (float*)d_out, (size_t)Nn * OUT, OUT - 1);
}
